// EventDenoisingMamba_21715354648666
// MI455X (gfx1250) — matrix operand order repaired and checked by reading, not compiled, not run on hardware
//
#include <hip/hip_runtime.h>
#include <hip/hip_bf16.h>
#include <math.h>

typedef __attribute__((ext_vector_type(16))) __bf16 v16bf;
typedef __attribute__((ext_vector_type(8)))  float  v8f;

#define NLAYERS 4
#define DMODEL  128
#define DSTATE  16
#define DCONV   4
#define DINNER  256
#define DTRANK  8
#define FEATD   16
#define BATCH   4
#define SEQLEN  8192
#define TOKENS  (BATCH * SEQLEN)
#define XPN     (DTRANK + 2 * DSTATE)   // 40

__device__ __forceinline__ float siluf(float x) { return x * (1.0f / (1.0f + __expf(-x))); }

union bfpk4 { __bf16 h[4]; uint2 u; };

// ---------------------------------------------------------------------------
// GEMM: C[M,N] = A[M,K] (fp32->bf16) x W[K,N] (fp32->bf16)
// Requirements: M % 64 == 0, K % 32 == 0, ldA/ldW multiples of 4.
// Block = 256 threads (8 waves), block tile 64(M) x 64(N).
// Wave w: n-tile = w&3, m-tiles = {2*(w>>2), 2*(w>>2)+1}  -> 2 wmma / K-step
// sharing one B fragment.
//
// LDS holds tiles in WMMA *fragment layout*: [tile][lane][16 bf16 contiguous],
// so each wave's fragment load is one contiguous 32B read (2x ds_load_b128)
// instead of 16 scalar gathers.
//   A frag (16x32 bf16): lane = 16*(k>>3 & 1) + row, elem e = (k&7) + 8*(k>>4)
//   B frag (32x16 bf16): lane = 16*(k>>3 & 1) + col, elem e = (k&7) + 8*(k>>4)
// A 4-aligned K-segment keeps lane fixed and e contiguous -> A staging stores
// one packed b64 per float4; B staging scatters 4 b16 (one per column).
// NGUARD=true adds N-edge guards (only the N=40 xproj GEMM needs them).
// ---------------------------------------------------------------------------
template <bool NGUARD>
__global__ __launch_bounds__(256) void gemm_bf16_wmma(
    const float* __restrict__ A, const float* __restrict__ W,
    float* __restrict__ C, int N, int K, int ldA, int ldW, int ldC)
{
    __shared__ __align__(32) __bf16 As[4 * 32 * 16];   // 4 m-tiles, frag layout
    __shared__ __align__(32) __bf16 Ws[4 * 32 * 16];   // 4 n-tiles, frag layout

    const int tid  = threadIdx.x;
    const int lane = tid & 31;
    const int wv   = tid >> 5;               // 0..7
    const int row0 = blockIdx.x * 64;
    const int colB = blockIdx.y * 64;

    const int mt   = (wv >> 2) * 2;          // first of two m-tiles (0 or 2)
    const int nt   = wv & 3;                 // n-tile
    const int col0 = colB + nt * 16;

    v8f acc0 = {}, acc1 = {};

    for (int k0 = 0; k0 < K; k0 += 32) {
        // --- stage A 64x32: 512 float4 segments, 2 per thread (branch-free).
        // seg holds k = seg*4 .. seg*4+3: half=(seg>>1)&1, e0=(seg&1)*4+8*(seg>>2)
        #pragma unroll
        for (int i = 0; i < 2; ++i) {
            int e  = tid * 2 + i;            // 0..511
            int r  = e >> 3, seg = e & 7;    // global row r, K segment
            const float4 f = *(const float4*)&A[(row0 + r) * ldA + k0 + seg * 4];
            bfpk4 pk;
            pk.h[0] = (__bf16)f.x; pk.h[1] = (__bf16)f.y;
            pk.h[2] = (__bf16)f.z; pk.h[3] = (__bf16)f.w;
            int alane = ((seg >> 1) & 1) * 16 + (r & 15);
            int e0    = (seg & 1) * 4 + (seg >> 2) * 8;
            *(uint2*)&As[(((r >> 4) * 32) + alane) * 16 + e0] = pk.u;
        }
        // --- stage W 32x64: 512 float4 segments, 2 per thread;
        // scatter 4 bf16 into fragment layout (lane varies with column).
        #pragma unroll
        for (int i = 0; i < 2; ++i) {
            int e = tid * 2 + i;
            int r = e >> 4, seg = e & 15;    // k row r (0..31), 4-col segment
            int c = colB + seg * 4;
            bfpk4 pk;
            if (!NGUARD || c + 3 < N) {
                const float4 f = *(const float4*)&W[(k0 + r) * ldW + c];
                pk.h[0] = (__bf16)f.x; pk.h[1] = (__bf16)f.y;
                pk.h[2] = (__bf16)f.z; pk.h[3] = (__bf16)f.w;
            } else {
                #pragma unroll
                for (int q = 0; q < 4; ++q)
                    pk.h[q] = (__bf16)((c + q < N) ? W[(k0 + r) * ldW + c + q] : 0.0f);
            }
            #pragma unroll
            for (int q = 0; q < 4; ++q) {
                int n     = seg * 4 + q;                       // 0..63
                int blane = ((r >> 3) & 1) * 16 + (n & 15);
                Ws[(((n >> 4) * 32) + blane) * 16 + (r & 7) + 8 * (r >> 4)] = pk.h[q];
            }
        }
        __syncthreads();

        if (k0 + 32 < K)   // speculative prefetch of next A tile -> global_prefetch_b8
            __builtin_prefetch(&A[(row0 + (tid >> 2)) * ldA + k0 + 32], 0, 1);

        // --- fragment loads: one contiguous 32B read per tile
        v16bf af0 = *(const v16bf*)&As[((mt * 32) + lane) * 16];
        v16bf af1 = *(const v16bf*)&As[(((mt + 1) * 32) + lane) * 16];
        v16bf bfr = *(const v16bf*)&Ws[((nt * 32) + lane) * 16];

        acc0 = __builtin_amdgcn_wmma_f32_16x16x32_bf16(
            false, af0, false, bfr, (short)0, acc0, false, false);
        acc1 = __builtin_amdgcn_wmma_f32_16x16x32_bf16(
            false, af1, false, bfr, (short)0, acc1, false, false);
        __syncthreads();
    }

    // --- store: VGPR r -> row = r + 8*(lane>>4), col = lane&15
    const int ccol = col0 + (lane & 15);
    if (!NGUARD || ccol < N) {
        #pragma unroll
        for (int r = 0; r < 8; ++r) {
            int rowa = row0 + mt * 16 + r + 8 * (lane >> 4);
            C[rowa * ldC + ccol]        = acc0[r];
            C[(rowa + 16) * ldC + ccol] = acc1[r];
        }
    }
}

// ---------------------------------------------------------------------------
// Embedding: x = features @ emb_W + emb_b   (K=16, bandwidth-trivial)
// ---------------------------------------------------------------------------
__global__ __launch_bounds__(256) void embed_kernel(
    const float* __restrict__ feat, const float* __restrict__ eW,
    const float* __restrict__ eb, float* __restrict__ x)
{
    int idx = blockIdx.x * blockDim.x + threadIdx.x;
    if (idx >= TOKENS * DMODEL) return;
    int c = idx & (DMODEL - 1);
    int t = idx >> 7;
    float s = eb[c];
    #pragma unroll
    for (int f = 0; f < FEATD; ++f)
        s += feat[t * FEATD + f] * eW[f * DMODEL + c];
    x[idx] = s;
}

// ---------------------------------------------------------------------------
// Causal depthwise conv1d (width 4) + bias + SiLU.  xi lives in xz cols 0..255.
// ---------------------------------------------------------------------------
__global__ __launch_bounds__(256) void conv_silu_kernel(
    const float* __restrict__ xz, const float* __restrict__ cw,
    const float* __restrict__ cb, float* __restrict__ xc)
{
    int idx = blockIdx.x * blockDim.x + threadIdx.x;
    if (idx >= TOKENS * DINNER) return;
    int d = idx & (DINNER - 1);
    int t = idx >> 8;
    int b = t / SEQLEN, l = t - b * SEQLEN;
    float s = cb[d];
    #pragma unroll
    for (int j = 0; j < DCONV; ++j) {
        int ls = l - (DCONV - 1) + j;
        if (ls >= 0) s += xz[(b * SEQLEN + ls) * (2 * DINNER) + d] * cw[d * DCONV + j];
    }
    xc[idx] = siluf(s);
}

// ---------------------------------------------------------------------------
// dt = softplus(x_dbl[:, :8] @ dt_W + dt_b)
// ---------------------------------------------------------------------------
__global__ __launch_bounds__(256) void dt_softplus_kernel(
    const float* __restrict__ xdbl, const float* __restrict__ dtW,
    const float* __restrict__ dtb, float* __restrict__ dt)
{
    int idx = blockIdx.x * blockDim.x + threadIdx.x;
    if (idx >= TOKENS * DINNER) return;
    int d = idx & (DINNER - 1);
    int t = idx >> 8;
    float s = dtb[d];
    #pragma unroll
    for (int r = 0; r < DTRANK; ++r)
        s += xdbl[t * XPN + r] * dtW[r * DINNER + d];
    dt[idx] = (s > 20.0f) ? s : log1pf(__expf(s));
}

// ---------------------------------------------------------------------------
// Selective scan. One 16-lane group per (b, d) channel; lane s holds state s.
// ---------------------------------------------------------------------------
__global__ __launch_bounds__(256) void scan_kernel(
    const float* __restrict__ dt, const float* __restrict__ xdbl,
    const float* __restrict__ xc, const float* __restrict__ A_log,
    float* __restrict__ ys)
{
    const int s = threadIdx.x & 15;
    const int g = blockIdx.x * 16 + (threadIdx.x >> 4);   // 0..1023 = b*256 + d
    const int b = g >> 8;
    const int d = g & (DINNER - 1);

    const float a = -__expf(A_log[d * DSTATE + s]);
    float h = 0.0f;
    const float* xb = xdbl + (long)b * SEQLEN * XPN;

    for (int l = 0; l < SEQLEN; ++l) {
        int t = b * SEQLEN + l;
        float dtv = dt[t * DINNER + d];
        float xv  = xc[t * DINNER + d];
        float Bv  = xb[l * XPN + DTRANK + s];
        float Cv  = xb[l * XPN + DTRANK + DSTATE + s];
        h = h * __expf(dtv * a) + dtv * Bv * xv;
        float p = h * Cv;
        p += __shfl_xor(p, 1, 16);
        p += __shfl_xor(p, 2, 16);
        p += __shfl_xor(p, 4, 16);
        p += __shfl_xor(p, 8, 16);
        if (s == 0) ys[t * DINNER + d] = p;
    }
}

// ---------------------------------------------------------------------------
// y = (ys + xc * Dp) * silu(z)   (z lives in xz cols 256..511)
// ---------------------------------------------------------------------------
__global__ __launch_bounds__(256) void gate_kernel(
    float* __restrict__ ys, const float* __restrict__ xc,
    const float* __restrict__ xz, const float* __restrict__ Dp)
{
    int idx = blockIdx.x * blockDim.x + threadIdx.x;
    if (idx >= TOKENS * DINNER) return;
    int d = idx & (DINNER - 1);
    int t = idx >> 8;
    float zv = xz[t * (2 * DINNER) + DINNER + d];
    float y  = ys[idx] + xc[idx] * Dp[d];
    ys[idx]  = y * siluf(zv);
}

// ---------------------------------------------------------------------------
// Head: out[t] = sigmoid(x[t,:] . head_W + head_b). One wave per token.
// ---------------------------------------------------------------------------
__global__ __launch_bounds__(256) void head_kernel(
    const float* __restrict__ x, const float* __restrict__ hW,
    const float* __restrict__ hb, float* __restrict__ out)
{
    int lane = threadIdx.x & 31;
    int t    = blockIdx.x * 8 + (threadIdx.x >> 5);
    if (t >= TOKENS) return;
    float s = 0.0f;
    #pragma unroll
    for (int j = 0; j < 4; ++j) {
        int c = lane + 32 * j;
        s += x[t * DMODEL + c] * hW[c];
    }
    s += __shfl_xor(s, 1, 32);
    s += __shfl_xor(s, 2, 32);
    s += __shfl_xor(s, 4, 32);
    s += __shfl_xor(s, 8, 32);
    s += __shfl_xor(s, 16, 32);
    if (lane == 0) out[t] = 1.0f / (1.0f + __expf(-(s + hb[0])));
}

// ---------------------------------------------------------------------------
extern "C" void kernel_launch(void* const* d_in, const int* in_sizes, int n_in,
                              void* d_out, int out_size, void* d_ws, size_t ws_size,
                              hipStream_t stream)
{
    const float* features = (const float*)d_in[0];
    const float* emb_W    = (const float*)d_in[1];
    const float* emb_b    = (const float*)d_in[2];
    const float* in_W     = (const float*)d_in[3];
    const float* conv_w   = (const float*)d_in[4];
    const float* conv_b   = (const float*)d_in[5];
    const float* xproj_W  = (const float*)d_in[6];
    const float* dt_W     = (const float*)d_in[7];
    const float* dt_b     = (const float*)d_in[8];
    const float* A_log    = (const float*)d_in[9];
    const float* Dp       = (const float*)d_in[10];
    const float* out_W    = (const float*)d_in[11];
    const float* head_W   = (const float*)d_in[12];
    const float* head_b   = (const float*)d_in[13];
    float* out            = (float*)d_out;

    // workspace carve-up (fp32)
    float* ws   = (float*)d_ws;
    float* x    = ws;  ws += (size_t)TOKENS * DMODEL;       // [T,128]
    float* xz   = ws;  ws += (size_t)TOKENS * 2 * DINNER;   // [T,512]
    float* xc   = ws;  ws += (size_t)TOKENS * DINNER;       // [T,256]
    float* xdbl = ws;  ws += (size_t)TOKENS * XPN;          // [T,40]
    float* dtb  = ws;  ws += (size_t)TOKENS * DINNER;       // [T,256]
    float* ys   = ws;  ws += (size_t)TOKENS * DINNER;       // [T,256]

    const int MT64 = TOKENS / 64;             // 512 M-tiles
    const int EW   = TOKENS * DINNER / 256;   // elementwise grid (32768)

    // embedding (K=16): simple bandwidth kernel
    embed_kernel<<<TOKENS * DMODEL / 256, 256, 0, stream>>>(features, emb_W, emb_b, x);

    for (int i = 0; i < NLAYERS; ++i) {
        const float* inW  = in_W    + (size_t)i * DMODEL * 2 * DINNER;
        const float* cw   = conv_w  + (size_t)i * DINNER * DCONV;
        const float* cb   = conv_b  + (size_t)i * DINNER;
        const float* xpW  = xproj_W + (size_t)i * DINNER * XPN;
        const float* dtW  = dt_W    + (size_t)i * DTRANK * DINNER;
        const float* dtbi = dt_b    + (size_t)i * DINNER;
        const float* Al   = A_log   + (size_t)i * DINNER * DSTATE;
        const float* Dpi  = Dp      + (size_t)i * DINNER;
        const float* oW   = out_W   + (size_t)i * DINNER * DMODEL;

        // xz = x @ in_W     (N=512, K=128) — no guards
        gemm_bf16_wmma<false><<<dim3(MT64, 8), 256, 0, stream>>>(
            x, inW, xz, 2 * DINNER, DMODEL, DMODEL, 2 * DINNER, 2 * DINNER);

        // xc = silu(conv1d(xi) + b)
        conv_silu_kernel<<<EW, 256, 0, stream>>>(xz, cw, cb, xc);

        // x_dbl = xc @ xproj_W   (N=40, K=256) — N guards
        gemm_bf16_wmma<true><<<dim3(MT64, 1), 256, 0, stream>>>(
            xc, xpW, xdbl, XPN, DINNER, DINNER, XPN, XPN);

        // dt = softplus(dt_raw @ dt_W + dt_b)
        dt_softplus_kernel<<<EW, 256, 0, stream>>>(xdbl, dtW, dtbi, dtb);

        // selective scan: 1024 channels, 16 per block
        scan_kernel<<<BATCH * DINNER / 16, 256, 0, stream>>>(dtb, xdbl, xc, Al, ys);

        // ys = (ys + xc*Dp) * silu(z)
        gate_kernel<<<EW, 256, 0, stream>>>(ys, xc, xz, Dpi);

        // x = ys @ out_W   (N=128, K=256) — no guards
        gemm_bf16_wmma<false><<<dim3(MT64, 2), 256, 0, stream>>>(
            ys, oW, x, DMODEL, DINNER, DINNER, DMODEL, DMODEL);
    }

    // logits -> sigmoid
    head_kernel<<<TOKENS / 8, 256, 0, stream>>>(x, head_W, head_b, out);
}
